// PointNetAllNetwork_7516192768116
// MI455X (gfx1250) — matrix operand order repaired and checked by reading, not compiled, not run on hardware
//
#include <hip/hip_runtime.h>
#include <hip/hip_bf16.h>
#include <math.h>

// ---------------------------------------------------------------------------
// Problem constants (from reference)
// ---------------------------------------------------------------------------
#define BDIM    4
#define NATOMS  256
#define MUL     32
#define HW      128     // radial MLP hidden width
#define NB      32      // num basis
#define CHUNK   32      // b-rows processed per pipeline pass (2 M-tiles)

#define LOG2F_    0.6931471805599453f
#define LOG2E_    1.4426950408889634f
#define Y0_       0.28209479177387814f   // 1/(2 sqrt(pi))
#define STEP_     (10.0f / 31.0f)
#define INVSTEP_  (31.0f / 10.0f)
#define INV_SQRT_N 0.0625f               // 1/sqrt(256)

typedef __attribute__((ext_vector_type(16))) __bf16 v16bf;
typedef __attribute__((ext_vector_type(8)))  __bf16 v8bf;
typedef __attribute__((ext_vector_type(8)))  float  v8f;

// ---------------------------------------------------------------------------
// Branch-free softplus(beta=5) on raw hardware transcendentals (base-2):
//   softplus(y) = max(y,0) + ln2 * log2(1 + 2^(-|y|*log2e))
// Argument of log2 is in [1,2] -> no denormal guard needed; lowers to a
// v_exp_f32 + v_log_f32 + a few dual-issue VALU ops, no EXEC divergence.
// ---------------------------------------------------------------------------
__device__ __forceinline__ float softplus5(float x) {
    float y = 5.0f * x;
    float t = __builtin_amdgcn_exp2f(-fabsf(y) * LOG2E_);
    return fmaxf(y, 0.0f) + LOG2F_ * __builtin_amdgcn_logf(1.0f + t);
}
__device__ __forceinline__ float ssp_act(float x) {   // ShiftedSoftplus
    return (softplus5(x) - LOG2F_) * 0.2f;
}
__device__ __forceinline__ float sp_act(float x) {    // Softplus
    return softplus5(x) * 0.2f;
}

// ---------------------------------------------------------------------------
// WMMA fragment helpers (wave32, v_wmma_f32_16x16x32_bf16).
// A-matrix (16x32 bf16, row-major in LDS, leading dim = ldk):
//   lane<16 : M=lane,    K = {0..7, 16..23}  (+ kt*32)
//   lane>=16: M=lane-16, K = {8..15, 24..31} (+ kt*32)
// ---------------------------------------------------------------------------
__device__ __forceinline__ v16bf load_a_frag(const __bf16* base, int row0,
                                             int ldk, int kt, int lane) {
    int m  = lane & 15;
    int k0 = kt * 32 + ((lane >> 4) << 3);          // 0 or 8
    const __bf16* p = base + (row0 + m) * ldk + k0;
    v8bf lo = *(const v8bf*)(p);
    v8bf hi = *(const v8bf*)(p + 16);
    return __builtin_shufflevector(lo, hi, 0,1,2,3,4,5,6,7,8,9,10,11,12,13,14,15);
}

// B-matrix fragments come pre-swizzled in global memory:
//   block (nt,kt): 32 lanes x 16 contiguous bf16 each;
//   lane: n = nt*16 + (lane&15), k = kt*32 + (lane>=16 ? 8:0) + (t&7) + (t>=8 ? 16:0) (t=0..15)
__device__ __forceinline__ v16bf load_b_frag(const __bf16* w, int nt, int kt,
                                             int numKt, int lane) {
    return *(const v16bf*)(w + (((nt * numKt + kt) * 32) + lane) * 16);
}

// D layout: VGPR r -> row = row0 + r + (lane>=16 ? 8:0), col = col0 + (lane&15)
__device__ __forceinline__ void store_d_ssp(__bf16* dst, int row0, int col0,
                                            int ldn, int lane, v8f c) {
    int m0 = row0 + ((lane >> 4) << 3);
    int n  = col0 + (lane & 15);
#pragma unroll
    for (int r = 0; r < 8; ++r)
        dst[(m0 + r) * ldn + n] = (__bf16)ssp_act(c[r]);
}

// ---------------------------------------------------------------------------
// Weight pre-swizzle: fold 1/sqrt(fan_in) (and Y0 for WO) into bf16 copies
// laid out fragment-ready for WMMA B operands.
// ---------------------------------------------------------------------------
__global__ void swizzle_kernel(const float* __restrict__ src, __bf16* __restrict__ dst,
                               int Ktot, int Ntot, float scale) {
    int idx   = blockIdx.x * 256 + threadIdx.x;
    int total = Ktot * Ntot;
    if (idx >= total) return;
    int t     = idx & 15;
    int lane  = (idx >> 4) & 31;
    int rest  = idx >> 9;
    int numKt = Ktot >> 5;
    int kt    = rest % numKt;
    int nt    = rest / numKt;
    int k     = kt * 32 + ((lane >> 4) << 3) + (t & 7) + ((t >> 3) << 4);
    int n     = nt * 16 + (lane & 15);
    dst[idx]  = (__bf16)(src[k * Ntot + n] * scale);
}

// Embedding lookup: f0[z,n,:] = emb_table[features[z,n,0]]
__global__ void embed_kernel(const int* __restrict__ feats,
                             const float* __restrict__ emb,
                             float* __restrict__ f0) {
    int idx = blockIdx.x * 256 + threadIdx.x;     // over B*N
    if (idx >= BDIM * NATOMS) return;
    int fv = feats[idx];
#pragma unroll
    for (int j = 0; j < MUL; ++j)
        f0[idx * MUL + j] = emb[fv * MUL + j];
}

// ---------------------------------------------------------------------------
// Fused layer kernel: one workgroup per (z, a).
//   basis(r_ab) -> MLP(32->128->128->1024) via bf16 WMMA -> contract with
//   fs[b,i] = f[z,b,i]*mask[z,b]/sqrt(N) -> out[j] -> f_out = sp(out)*mask[a]
// Per-wave tile assignment: GEMM1/GEMM2 tiles all have nt == wid, so their
// weight (B) fragments are loaded ONCE per workgroup and live in registers.
// ---------------------------------------------------------------------------
__global__ __launch_bounds__(256)
void layer_kernel(const float* __restrict__ geo,  const float* __restrict__ mask,
                  const float* __restrict__ f_in, float* __restrict__ f_out,
                  const __bf16* __restrict__ W0s, const __bf16* __restrict__ W1s,
                  const __bf16* __restrict__ WOs) {
    __shared__ __attribute__((aligned(32))) float  fs[NATOMS * MUL];   // 32 KB
    __shared__ __attribute__((aligned(32))) __bf16 X0[CHUNK * NB];     //  2 KB
    __shared__ __attribute__((aligned(32))) __bf16 H1s[CHUNK * HW];    //  8 KB
    __shared__ __attribute__((aligned(32))) __bf16 H2s[CHUNK * HW];    //  8 KB
    __shared__ float outacc[MUL];
    __shared__ float rbuf[CHUNK];

    const int z = blockIdx.y, a = blockIdx.x;
    const int tid = threadIdx.x, lane = tid & 31, wid = tid >> 5;

    // Chunk-invariant weight fragments (this wave's fixed nt == wid).
    v16bf bW0 = load_b_frag(W0s, wid, 0, 1, lane);
    v16bf bW1[4];
#pragma unroll
    for (int kt = 0; kt < 4; ++kt) bW1[kt] = load_b_frag(W1s, wid, kt, 4, lane);

    // fs: masked, pre-scaled features of all source atoms
    {
        int   b  = tid;
        float mb = mask[z * NATOMS + b] * INV_SQRT_N;
#pragma unroll 8
        for (int i = 0; i < MUL; ++i)
            fs[b * MUL + i] = f_in[(z * NATOMS + b) * MUL + i] * mb;
    }
    if (tid < MUL) outacc[tid] = 0.0f;

    const float gax = geo[(z * NATOMS + a) * 3 + 0];
    const float gay = geo[(z * NATOMS + a) * 3 + 1];
    const float gaz = geo[(z * NATOMS + a) * 3 + 2];
    __syncthreads();

    for (int chunk = 0; chunk < NATOMS / CHUNK; ++chunk) {
        const int b0 = chunk * CHUNK;
        if (chunk) __syncthreads();               // LDS reuse fence

        if (tid < CHUNK) {
            int   b  = b0 + tid;
            float dx = geo[(z * NATOMS + b) * 3 + 0] - gax;
            float dy = geo[(z * NATOMS + b) * 3 + 1] - gay;
            float dz = geo[(z * NATOMS + b) * 3 + 2] - gaz;
            rbuf[tid] = sqrtf(dx * dx + dy * dy + dz * dz);
        }
        __syncthreads();

        // Cosine-bump radial basis -> bf16 A tile [CHUNK,32] (branch-free)
#pragma unroll
        for (int e4 = 0; e4 < 4; ++e4) {
            int   e   = e4 * 256 + tid;
            int   row = e >> 5, c = e & 31;
            float u   = (rbuf[row] - (float)c * STEP_) * INVSTEP_;
            float cv  = __cosf(1.57079632679f * u);
            float v   = (fabsf(u) < 1.0f) ? cv * cv : 0.0f;   // v_cndmask
            X0[e] = (__bf16)v;
        }
        __syncthreads();

        // GEMM1: [CHUNK,32] x [32,128]; this wave: tiles (mt, nt=wid), K=32
#pragma unroll
        for (int mt = 0; mt < 2; ++mt) {
            v8f  c  = {};
            v16bf af = load_a_frag(X0, mt * 16, NB, 0, lane);
            c = __builtin_amdgcn_wmma_f32_16x16x32_bf16(false, af, false, bW0,
                                                        (short)0, c, false, false);
            store_d_ssp(H1s, mt * 16, wid * 16, HW, lane, c);
        }
        __syncthreads();

        // GEMM2: [CHUNK,128] x [128,128]; tiles (mt, nt=wid), K=128
#pragma unroll
        for (int mt = 0; mt < 2; ++mt) {
            v16bf af[4];
#pragma unroll
            for (int kt = 0; kt < 4; ++kt)
                af[kt] = load_a_frag(H1s, mt * 16, HW, kt, lane);
            v8f c = {};
#pragma unroll
            for (int kt = 0; kt < 4; ++kt)
                c = __builtin_amdgcn_wmma_f32_16x16x32_bf16(false, af[kt], false, bW1[kt],
                                                            (short)0, c, false, false);
            store_d_ssp(H2s, mt * 16, wid * 16, HW, lane, c);
        }
        __syncthreads();

        // GEMM3: [CHUNK,128] x [128,1024], fused einsum contraction.
        // Tile (mt,nt): cols nt*16.. -> j = nt>>1 fixed, i = (nt&1)*16 + lane%16
#pragma unroll 2
        for (int it = 0; it < 16; ++it) {
            int t  = it * 8 + wid;
            int mt = t >> 6, nt = t & 63;
            if (it + 1 < 16) {  // gfx1250 global_prefetch of next B block
                int ntn = ((it + 1) * 8 + wid) & 63;
                __builtin_prefetch((const void*)(WOs + ((ntn * 4) * 32 + lane) * 16), 0, 1);
            }
            v16bf af[4], bf[4];
#pragma unroll
            for (int kt = 0; kt < 4; ++kt) {
                af[kt] = load_a_frag(H2s, mt * 16, HW, kt, lane);
                bf[kt] = load_b_frag(WOs, nt, kt, 4, lane);
            }
            v8f c = {};
#pragma unroll
            for (int kt = 0; kt < 4; ++kt)
                c = __builtin_amdgcn_wmma_f32_16x16x32_bf16(false, af[kt], false, bf[kt],
                                                            (short)0, c, false, false);

            int   j   = nt >> 1;
            int   i   = ((nt & 1) << 4) + (lane & 15);
            int   br0 = b0 + mt * 16 + ((lane >> 4) << 3);
            float s   = 0.0f;
#pragma unroll
            for (int r = 0; r < 8; ++r) s += c[r] * fs[(br0 + r) * MUL + i];
#pragma unroll
            for (int off = 16; off; off >>= 1) s += __shfl_xor(s, off);
            if (lane == 0) atomicAdd(&outacc[j], s);   // ds_add_f32
        }
    }
    __syncthreads();

    if (tid < MUL) {
        float ma = mask[z * NATOMS + a];
        f_out[(z * NATOMS + a) * MUL + tid] = sp_act(outacc[tid]) * ma;
    }
}

// ---------------------------------------------------------------------------
// Pointwise resnet (32 -> relu 64 -> 32) + masked mean aggregate over atoms.
// ---------------------------------------------------------------------------
__global__ __launch_bounds__(256)
void resnet_agg_kernel(const float* __restrict__ f, const float* __restrict__ mask,
                       const float* __restrict__ W1, const float* __restrict__ W2,
                       float* __restrict__ out) {
    __shared__ float acc[MUL];
    __shared__ float msum;
    int z = blockIdx.x, tid = threadIdx.x;
    if (tid < MUL) acc[tid] = 0.0f;
    if (tid == 0) msum = 0.0f;
    __syncthreads();

    float m = mask[z * NATOMS + tid];
    float fv[MUL];
#pragma unroll
    for (int i = 0; i < MUL; ++i) fv[i] = f[(z * NATOMS + tid) * MUL + i];

    float hb[2 * MUL];
#pragma unroll 4
    for (int j = 0; j < 2 * MUL; ++j) {
        float s = 0.0f;
#pragma unroll
        for (int i = 0; i < MUL; ++i) s += fv[i] * W1[i * (2 * MUL) + j];
        hb[j] = fmaxf(s, 0.0f);
    }
#pragma unroll 4
    for (int i = 0; i < MUL; ++i) {
        float s = 0.0f;
#pragma unroll
        for (int j = 0; j < 2 * MUL; ++j) s += hb[j] * W2[j * MUL + i];
        atomicAdd(&acc[i], (fv[i] + s) * m);
    }
    atomicAdd(&msum, m);
    __syncthreads();
    if (tid < MUL) out[z * MUL + tid] = acc[tid] / msum;
}

// ---------------------------------------------------------------------------
// Workspace layout (bytes)
// ---------------------------------------------------------------------------
#define OFF_W0  0                                   // 3 * 4096   bf16
#define OFF_W1  (OFF_W0 + 3 * 4096 * 2)             // 3 * 16384  bf16
#define OFF_WO  (OFF_W1 + 3 * 16384 * 2)            // 3 * 131072 bf16
#define OFF_FA  (OFF_WO + 3 * 131072 * 2)           // 4*256*32 f32
#define OFF_FB  (OFF_FA + BDIM * NATOMS * MUL * 4)  // 4*256*32 f32
// total ~1.15 MB

extern "C" void kernel_launch(void* const* d_in, const int* in_sizes, int n_in,
                              void* d_out, int out_size, void* d_ws, size_t ws_size,
                              hipStream_t stream) {
    (void)in_sizes; (void)n_in; (void)out_size; (void)ws_size;
    const int*   features = (const int*)d_in[0];
    const float* geometry = (const float*)d_in[1];
    const float* mask     = (const float*)d_in[2];
    const float* emb      = (const float*)d_in[3];
    const float* RW0      = (const float*)d_in[4];   // [3,32,128]
    const float* RW1      = (const float*)d_in[5];   // [3,128,128]
    const float* RWO      = (const float*)d_in[6];   // [3,128,1024]
    const float* RES_W1   = (const float*)d_in[7];   // [32,64]
    const float* RES_W2   = (const float*)d_in[8];   // [64,32]
    float*       out      = (float*)d_out;

    char*   ws  = (char*)d_ws;
    __bf16* W0s = (__bf16*)(ws + OFF_W0);
    __bf16* W1s = (__bf16*)(ws + OFF_W1);
    __bf16* WOs = (__bf16*)(ws + OFF_WO);
    float*  fA  = (float*)(ws + OFF_FA);
    float*  fB  = (float*)(ws + OFF_FB);

    const float s0 = 0.17677669529663687f;            // 1/sqrt(32)
    const float s1 = 0.08838834764831845f;            // 1/sqrt(128)
    const float so = Y0_ * 0.08838834764831845f;      // Y0/sqrt(128)

    for (int l = 0; l < 3; ++l) {
        swizzle_kernel<<<(4096   + 255) / 256, 256, 0, stream>>>(RW0 + l * 4096,   W0s + l * 4096,   32,  128,  s0);
        swizzle_kernel<<<(16384  + 255) / 256, 256, 0, stream>>>(RW1 + l * 16384,  W1s + l * 16384,  128, 128,  s1);
        swizzle_kernel<<<(131072 + 255) / 256, 256, 0, stream>>>(RWO + l * 131072, WOs + l * 131072, 128, 1024, so);
    }
    embed_kernel<<<(BDIM * NATOMS + 255) / 256, 256, 0, stream>>>(features, emb, fA);

    dim3 grid(NATOMS, BDIM);
    layer_kernel<<<grid, 256, 0, stream>>>(geometry, mask, fA, fB, W0s,          W1s,           WOs);
    layer_kernel<<<grid, 256, 0, stream>>>(geometry, mask, fB, fA, W0s + 4096,   W1s + 16384,   WOs + 131072);
    layer_kernel<<<grid, 256, 0, stream>>>(geometry, mask, fA, fB, W0s + 8192,   W1s + 32768,   WOs + 262144);

    resnet_agg_kernel<<<BDIM, 256, 0, stream>>>(fB, mask, RES_W1, RES_W2, out);
}
